// DiffTransformerBlock_16346645528771
// MI455X (gfx1250) — matrix operand order repaired and checked by reading, not compiled, not run on hardware
//
#include <hip/hip_runtime.h>
#include <hip/hip_bf16.h>

typedef _Float16 f16;
typedef __attribute__((ext_vector_type(16))) _Float16 v16h;
typedef __attribute__((ext_vector_type(8)))  _Float16 v8h;
typedef __attribute__((ext_vector_type(8)))  float    v8f;

#define NPIX  16384      // 128*128
#define NBAT  8
#define CDIM  192
#define HEADS 8
#define CF    24         // CDIM / HEADS
#define HID   510
#define HIDP  512        // HID padded to multiple of 32 (WMMA K step)
#define PINM  1024       // 2*HID=1020 padded to multiple of 16 (WMMA M tile)

// ---------------------------------------------------------------------------
// FiLM modulation vectors: m[b][o] = sum_i kv[b][i] * kern[o][i]   (O = 384)
// ---------------------------------------------------------------------------
__global__ void film_kernel(const float* __restrict__ kv,
                            const float* __restrict__ kern,
                            float* __restrict__ m, int O) {
    int idx = blockIdx.x * blockDim.x + threadIdx.x;
    if (idx >= NBAT * O) return;
    int b = idx / O, o = idx % O;
    const float* k = kv + b * 256;
    const float* w = kern + o * 256;
    float s = 0.f;
    #pragma unroll 4
    for (int i = 0; i < 256; ++i) s += k[i] * w[i];
    m[idx] = s;
}

// ---------------------------------------------------------------------------
// fp32 -> f16 weight repack with zero padding: dst[Mdst][Kdst] <- src[Msrc][Ksrc]
// ---------------------------------------------------------------------------
__global__ void w2h_kernel(const float* __restrict__ src, f16* __restrict__ dst,
                           int Msrc, int Ksrc, int Mdst, int Kdst) {
    long idx = (long)blockIdx.x * blockDim.x + threadIdx.x;
    if (idx >= (long)Mdst * Kdst) return;
    int mm = (int)(idx / Kdst), kk = (int)(idx % Kdst);
    float v = (mm < Msrc && kk < Ksrc) ? src[(long)mm * Ksrc + kk] : 0.f;
    dst[idx] = (f16)v;
}

// ---------------------------------------------------------------------------
// Fused per-pixel LayerNorm (over C=192) + FiLM, transposed f16 output:
//   in  x  : [8][192][NPIX] fp32 (NCHW)
//   out xn : [8][NPIX][192] f16  (channel-contiguous -> WMMA B-fragment loads)
// ---------------------------------------------------------------------------
__global__ void ln_film_kernel(const float* __restrict__ x,
                               const float* __restrict__ lw,
                               const float* __restrict__ lb,
                               const float* __restrict__ m,   // [8][2*192]
                               f16* __restrict__ xn) {
    long idx = (long)blockIdx.x * blockDim.x + threadIdx.x;
    if (idx >= (long)NBAT * NPIX) return;
    int b = (int)(idx / NPIX), pix = (int)(idx % NPIX);
    const float* xp = x + ((long)b * CDIM) * NPIX + pix;
    float s = 0.f, ss = 0.f;
    for (int c = 0; c < CDIM; ++c) {
        float v = xp[(long)c * NPIX];
        s += v; ss += v * v;
    }
    float mu   = s * (1.f / CDIM);
    float var  = ss * (1.f / CDIM) - mu * mu;
    float rinv = rsqrtf(var + 1e-5f);
    const float* mb = m + b * (2 * CDIM);
    f16* op = xn + idx * CDIM;
    for (int c = 0; c < CDIM; ++c) {
        float v = (xp[(long)c * NPIX] - mu) * rinv * lw[c] + lb[c];
        v = v * mb[c] + mb[CDIM + c];
        op[c] = (f16)v;
    }
}

// ---------------------------------------------------------------------------
// WMMA GEMM, 2x2 register-tiled, K fully unrolled at compile time:
//   Out[b][m][pix] (+ Res) = sum_k A[m][k] * Bmat[b*NPIX+pix][k]
//   A    : [M][K] f16 (row-major weights, K mult of 32, M mult of 32)
//   Bmat : [8*NPIX][K] f16 (pixel-major, channel-contiguous)
//   Out  : fp32, batch stride = out_ch_stride*NPIX, channel stride = NPIX
// Each wave owns a 32(M) x 32(N) block: 4 accumulators, A frags shared across
// the two N-tiles, B frags shared across the two M-tiles. Full unroll (K is a
// template arg: 192 -> 6 steps, 512 -> 16 steps) gives straight-line code: the
// scheduler hoists future fragment loads over current WMMAs with no register
// rotation moves and no in-loop branches (EXEC stays all-1s for WMMA).
// Fragment layouts per CDNA5 ISA 7.12.2 (16-bit A 16x32, B 32x16, f32 C 16x16).
// ---------------------------------------------------------------------------
template<int K>
__global__ __launch_bounds__(256)
void wmma_gemm_kernel(const f16* __restrict__ A, const f16* __restrict__ Bmat,
                      float* __restrict__ Out, const float* __restrict__ Res,
                      int M, int out_ch_stride, int res_ch_stride) {
    const int tpm = M >> 5;                       // M tile-pairs
    const int tpn = (NBAT * NPIX) >> 5;           // N tile-pairs
    int wave = (blockIdx.x << 3) + (threadIdx.x >> 5);
    if (wave >= tpm * tpn) return;                // wave-uniform guard
    int tn = wave / tpm;
    int tm = wave - tn * tpm;

    int lane = threadIdx.x & 31;
    int half = lane >> 4;                         // hi/lo 16-lane group
    int l16  = lane & 15;

    int  row0 = (tm << 5) + l16;                  // A rows for M-tile 0
    long p0   = ((long)tn << 5) + l16;            // pixel cols for N-tile 0
    // A 16x32 f16 frag: lanes 0-15 K{k0..k0+7,k0+16..k0+23}; lanes 16-31 +8
    const f16* A0 = A + (long)row0 * K + half * 8;
    const f16* A1 = A0 + (long)16 * K;
    // B 32x16 f16 frag: lane->col N, K{k0..k0+7,k0+16..k0+23}; lanes 16-31 +8
    const f16* B0 = Bmat + p0 * K + half * 8;
    const f16* B1 = B0 + (long)16 * K;

    v8f acc00 = {}, acc01 = {}, acc10 = {}, acc11 = {};
    #pragma unroll
    for (int k0 = 0; k0 < K; k0 += 32) {
        if (k0 + 64 <= K) {                        // folds away (K constexpr)
            __builtin_prefetch(B0 + k0 + 32, 0, 1);
            __builtin_prefetch(B1 + k0 + 32, 0, 1);
        }
        union { v16h v; v8h h[2]; } ta0, ta1;
        ta0.h[0] = *(const v8h*)(A0 + k0);
        ta0.h[1] = *(const v8h*)(A0 + k0 + 16);
        ta1.h[0] = *(const v8h*)(A1 + k0);
        ta1.h[1] = *(const v8h*)(A1 + k0 + 16);
        union { v16h v; v8h h[2]; } tb0, tb1;
        tb0.h[0] = *(const v8h*)(B0 + k0);
        tb0.h[1] = *(const v8h*)(B0 + k0 + 16);
        tb1.h[0] = *(const v8h*)(B1 + k0);
        tb1.h[1] = *(const v8h*)(B1 + k0 + 16);
        v16h b0 = tb0.v;
        v16h b1 = tb1.v;
        acc00 = __builtin_amdgcn_wmma_f32_16x16x32_f16(false, ta0.v, false, b0,
                                                       (short)0, acc00, false, false);
        acc01 = __builtin_amdgcn_wmma_f32_16x16x32_f16(false, ta0.v, false, b1,
                                                       (short)0, acc01, false, false);
        acc10 = __builtin_amdgcn_wmma_f32_16x16x32_f16(false, ta1.v, false, b0,
                                                       (short)0, acc10, false, false);
        acc11 = __builtin_amdgcn_wmma_f32_16x16x32_f16(false, ta1.v, false, b1,
                                                       (short)0, acc11, false, false);
    }

    // C layout per tile: lane<16 -> N=l16, M=base+r ; lane>=16 -> M=base+8+r
    int  bimg = (int)(p0 / NPIX);
    int  pix0 = (int)(p0 - (long)bimg * NPIX);    // N-pair never crosses batch
    long ob   = (long)bimg * out_ch_stride * NPIX;
    long rb   = (long)bimg * res_ch_stride * NPIX;
    int  m00  = (tm << 5) + (half << 3);          // rows of acc0x
    int  m10  = m00 + 16;                         // rows of acc1x
    #pragma unroll
    for (int r = 0; r < 8; ++r) {
        float v00 = acc00[r], v01 = acc01[r], v10 = acc10[r], v11 = acc11[r];
        if (Res) {
            v00 += Res[rb + (long)(m00 + r) * NPIX + pix0];
            v01 += Res[rb + (long)(m00 + r) * NPIX + pix0 + 16];
            v10 += Res[rb + (long)(m10 + r) * NPIX + pix0];
            v11 += Res[rb + (long)(m10 + r) * NPIX + pix0 + 16];
        }
        Out[ob + (long)(m00 + r) * NPIX + pix0]      = v00;
        Out[ob + (long)(m00 + r) * NPIX + pix0 + 16] = v01;
        Out[ob + (long)(m10 + r) * NPIX + pix0]      = v10;
        Out[ob + (long)(m10 + r) * NPIX + pix0 + 16] = v11;
    }
}

// ---------------------------------------------------------------------------
// Depthwise 3x3 SAME (zero pad), fp32 NCHW.  in batch stride = in_stride*NPIX.
// ---------------------------------------------------------------------------
__global__ void dw3x3_kernel(const float* __restrict__ in,
                             const float* __restrict__ w,  // [CH][1][3][3]
                             float* __restrict__ out, int CH, int in_stride) {
    long idx = (long)blockIdx.x * blockDim.x + threadIdx.x;
    if (idx >= (long)NBAT * CH * NPIX) return;
    int pix = (int)(idx % NPIX);
    int c   = (int)((idx / NPIX) % CH);
    int b   = (int)(idx / ((long)NPIX * CH));
    int y = pix >> 7, xx = pix & 127;
    const float* ip = in + ((long)b * in_stride + c) * NPIX;
    const float* wp = w + c * 9;
    float acc = 0.f;
    #pragma unroll
    for (int ky = 0; ky < 3; ++ky) {
        int yy = y + ky - 1;
        if (yy < 0 || yy > 127) continue;
        #pragma unroll
        for (int kx = 0; kx < 3; ++kx) {
            int xw = xx + kx - 1;
            if (xw < 0 || xw > 127) continue;
            acc += wp[ky * 3 + kx] * ip[yy * 128 + xw];
        }
    }
    out[((long)b * CH + c) * NPIX + pix] = acc;
}

// ---------------------------------------------------------------------------
// Double L2 normalization of q,k rows over N=16384, in place.
// One block per (b, c) with c in [0,384) of qkv buffer [8][576][NPIX].
// ---------------------------------------------------------------------------
__global__ __launch_bounds__(256)
void qknorm_kernel(float* __restrict__ qk) {
    int b = blockIdx.x / 384, c = blockIdx.x % 384;
    float* row = qk + ((long)b * 576 + c) * NPIX;
    __shared__ float red[256];
    float ss = 0.f;
    for (int i = threadIdx.x; i < NPIX; i += 256) { float v = row[i]; ss += v * v; }
    red[threadIdx.x] = ss;
    __syncthreads();
    for (int s = 128; s > 0; s >>= 1) {
        if (threadIdx.x < s) red[threadIdx.x] += red[threadIdx.x + s];
        __syncthreads();
    }
    float n  = sqrtf(red[0]);
    float n1 = n + 1e-12f;
    float n2 = n / n1;
    float scale = 1.f / (n1 * fmaxf(n2, 1e-12f));
    for (int i = threadIdx.x; i < NPIX; i += 256) row[i] *= scale;
}

// ---------------------------------------------------------------------------
// Channel attention per (b, head): S = q k^T * temp -> softmax -> out = S v.
// qkv: [8][576][NPIX] (q:0..191, k:192..383, v:384..575, already normalized).
// out: [8][NPIX][192] f16 (transposed for the proj WMMA GEMM B layout).
// 24x24 attention lives in LDS; only ~2% of total FLOPs -> VALU is fine.
// ---------------------------------------------------------------------------
__global__ __launch_bounds__(256)
void attn_kernel(const float* __restrict__ qkv,
                 const float* __restrict__ temp,   // [HEADS]
                 f16* __restrict__ out) {
    int b  = blockIdx.x >> 3;
    int hd = blockIdx.x & 7;
    const float* qb = qkv + ((long)b * 576 + hd * CF) * NPIX;
    const float* kb = qb + (long)CDIM * NPIX;
    const float* vb = qb + (long)(2 * CDIM) * NPIX;
    __shared__ float sA[CF * CF];

    // phase 1: 576 dot products of length 16384
    for (int e = threadIdx.x; e < CF * CF; e += 256) {
        int c = e / CF, d = e % CF;
        const float* qr = qb + (long)c * NPIX;
        const float* kr = kb + (long)d * NPIX;
        float s = 0.f;
        #pragma unroll 4
        for (int i = 0; i < NPIX; ++i) s += qr[i] * kr[i];
        sA[e] = s;
    }
    __syncthreads();

    // phase 2: row softmax (temperature applied pre-softmax)
    if (threadIdx.x < CF) {
        float t = temp[hd];
        float* rowp = sA + threadIdx.x * CF;
        float mx = -3.0e38f;
        for (int d = 0; d < CF; ++d) { rowp[d] *= t; mx = fmaxf(mx, rowp[d]); }
        float sum = 0.f;
        for (int d = 0; d < CF; ++d) { float e = __expf(rowp[d] - mx); rowp[d] = e; sum += e; }
        float inv = 1.f / sum;
        for (int d = 0; d < CF; ++d) rowp[d] *= inv;
    }
    __syncthreads();

    // phase 3: out[c][pix] = sum_d attn[c][d] * v[d][pix], stored transposed f16
    for (int pix = threadIdx.x; pix < NPIX; pix += 256) {
        float vv[CF];
        #pragma unroll
        for (int d = 0; d < CF; ++d) vv[d] = vb[(long)d * NPIX + pix];
        f16* op = out + ((long)b * NPIX + pix) * CDIM + hd * CF;
        #pragma unroll
        for (int c = 0; c < CF; ++c) {
            const float* ar = sA + c * CF;
            float acc = 0.f;
            #pragma unroll
            for (int d = 0; d < CF; ++d) acc += ar[d] * vv[d];
            op[c] = (f16)acc;
        }
    }
}

// ---------------------------------------------------------------------------
// FFN depthwise 3x3 + exact-GeLU gate, fused; writes padded transposed f16:
//   in  u : [8][PINM][NPIX] fp32 (channels 0..1019 valid)
//   out   : [8][NPIX][HIDP] f16, out[..][c] = gelu(dw(u1)_c) * dw(u2)_c,
//           channels 510,511 zero-padded (so pout WMMA K=512 is exact).
// ---------------------------------------------------------------------------
__global__ void dwgate_kernel(const float* __restrict__ u,
                              const float* __restrict__ w,   // [1020][9]
                              f16* __restrict__ out) {
    long idx = (long)blockIdx.x * blockDim.x + threadIdx.x;
    if (idx >= (long)NBAT * NPIX * HIDP) return;
    int  c   = (int)(idx % HIDP);
    long bp  = idx / HIDP;
    int  pix = (int)(bp % NPIX);
    int  b   = (int)(bp / NPIX);
    if (c >= HID) { out[idx] = (f16)0.f; return; }
    int y = pix >> 7, xx = pix & 127;
    const float* i1 = u + ((long)b * PINM + c) * NPIX;
    const float* i2 = i1 + (long)HID * NPIX;
    const float* w1 = w + c * 9;
    const float* w2 = w + (c + HID) * 9;
    float a1 = 0.f, a2 = 0.f;
    #pragma unroll
    for (int ky = 0; ky < 3; ++ky) {
        int yy = y + ky - 1;
        if (yy < 0 || yy > 127) continue;
        #pragma unroll
        for (int kx = 0; kx < 3; ++kx) {
            int xw = xx + kx - 1;
            if (xw < 0 || xw > 127) continue;
            int o = yy * 128 + xw, kk = ky * 3 + kx;
            a1 += w1[kk] * i1[o];
            a2 += w2[kk] * i2[o];
        }
    }
    float g = 0.5f * a1 * (1.f + erff(a1 * 0.70710678118654752f)); // exact GeLU
    out[idx] = (f16)(g * a2);
}

// ---------------------------------------------------------------------------
extern "C" void kernel_launch(void* const* d_in, const int* in_sizes, int n_in,
                              void* d_out, int out_size, void* d_ws, size_t ws_size,
                              hipStream_t stream) {
    const float* x         = (const float*)d_in[0];
    const float* k_v       = (const float*)d_in[1];
    const float* ln1_w     = (const float*)d_in[2];
    const float* ln1_b     = (const float*)d_in[3];
    const float* attn_kern = (const float*)d_in[4];
    const float* qkv_w     = (const float*)d_in[5];
    const float* qkv_dw_w  = (const float*)d_in[6];
    const float* proj_w    = (const float*)d_in[7];
    const float* temp      = (const float*)d_in[8];
    const float* ln2_w     = (const float*)d_in[9];
    const float* ln2_b     = (const float*)d_in[10];
    const float* ffn_kern  = (const float*)d_in[11];
    const float* pin_w     = (const float*)d_in[12];
    const float* ffn_dw_w  = (const float*)d_in[13];
    const float* pout_w    = (const float*)d_in[14];

    char* ws = (char*)d_ws;
    size_t off = 0;
    auto alloc = [&](size_t bytes) {
        void* p = ws + off;
        off = (off + bytes + 255) & ~(size_t)255;
        return p;
    };
    f16*   wq16     = (f16*)  alloc((size_t)3 * CDIM * CDIM * sizeof(f16));      // 576x192
    f16*   wproj16  = (f16*)  alloc((size_t)CDIM * CDIM * sizeof(f16));          // 192x192
    f16*   wpin16   = (f16*)  alloc((size_t)PINM * CDIM * sizeof(f16));          // 1024x192
    f16*   wpout16  = (f16*)  alloc((size_t)CDIM * HIDP * sizeof(f16));          // 192x512
    float* m_attn   = (float*)alloc((size_t)NBAT * 2 * CDIM * sizeof(float));
    float* m_ffn    = (float*)alloc((size_t)NBAT * 2 * CDIM * sizeof(float));
    f16*   xn       = (f16*)  alloc((size_t)NBAT * NPIX * CDIM * sizeof(f16));   // also attn_out
    float* gemm_buf = (float*)alloc((size_t)NBAT * PINM * NPIX * sizeof(float)); // qkv_pre / pin_pre
    float* dwbuf    = (float*)alloc((size_t)NBAT * 3 * CDIM * NPIX * sizeof(float)); // qkv / u(f16)
    float* xres     = (float*)alloc((size_t)NBAT * CDIM * NPIX * sizeof(float));

    const int TB = 256;
    // --- weight repacks to f16 (with padding for pin M and pout K) ---
    w2h_kernel<<<(3*CDIM*CDIM + TB-1)/TB, TB, 0, stream>>>(qkv_w,  wq16,   3*CDIM, CDIM, 3*CDIM, CDIM);
    w2h_kernel<<<(CDIM*CDIM   + TB-1)/TB, TB, 0, stream>>>(proj_w, wproj16, CDIM,  CDIM, CDIM,  CDIM);
    w2h_kernel<<<(PINM*CDIM   + TB-1)/TB, TB, 0, stream>>>(pin_w,  wpin16, 2*HID,  CDIM, PINM,  CDIM);
    w2h_kernel<<<(CDIM*HIDP   + TB-1)/TB, TB, 0, stream>>>(pout_w, wpout16, CDIM,  HID,  CDIM,  HIDP);

    // --- FiLM vectors ---
    film_kernel<<<(NBAT*2*CDIM + TB-1)/TB, TB, 0, stream>>>(k_v, attn_kern, m_attn, 2*CDIM);
    film_kernel<<<(NBAT*2*CDIM + TB-1)/TB, TB, 0, stream>>>(k_v, ffn_kern,  m_ffn,  2*CDIM);

    const long npx = (long)NBAT * NPIX;

    // ===== attention branch =====
    ln_film_kernel<<<(int)((npx + TB-1)/TB), TB, 0, stream>>>(x, ln1_w, ln1_b, m_attn, xn);

    // qkv 1x1 conv: [576,192] x [192, 8*16384] (WMMA, 32x32 per wave)
    {
        int waves = ((3*CDIM) >> 5) * (int)(npx >> 5);
        wmma_gemm_kernel<CDIM><<<waves/8, TB, 0, stream>>>(wq16, xn, gemm_buf, nullptr,
                                                           3*CDIM, 3*CDIM, 0);
    }
    dw3x3_kernel<<<(int)((npx*3*CDIM)/TB), TB, 0, stream>>>(gemm_buf, qkv_dw_w, dwbuf,
                                                            3*CDIM, 3*CDIM);
    qknorm_kernel<<<NBAT*2*CDIM, TB, 0, stream>>>(dwbuf);
    attn_kernel<<<NBAT*HEADS, TB, 0, stream>>>(dwbuf, temp, xn /* attn_out reuse */);

    // proj 1x1 + residual: xres = x + Wproj @ attn_out (WMMA)
    {
        int waves = (CDIM >> 5) * (int)(npx >> 5);
        wmma_gemm_kernel<CDIM><<<waves/8, TB, 0, stream>>>(wproj16, xn, xres, x,
                                                           CDIM, CDIM, CDIM);
    }

    // ===== FFN branch =====
    ln_film_kernel<<<(int)((npx + TB-1)/TB), TB, 0, stream>>>(xres, ln2_w, ln2_b, m_ffn, xn);

    // pin 1x1: [1024,192] x [192, 8*16384] (WMMA; rows 1020..1023 are zero pad)
    {
        int waves = (PINM >> 5) * (int)(npx >> 5);
        wmma_gemm_kernel<CDIM><<<waves/8, TB, 0, stream>>>(wpin16, xn, gemm_buf, nullptr,
                                                           PINM, PINM, 0);
    }

    // depthwise 3x3 + GeLU gate -> transposed padded f16 [8][NPIX][512]
    f16* u16 = (f16*)dwbuf;  // reuse (134 MB <= 302 MB)
    dwgate_kernel<<<(int)((npx*HIDP)/TB), TB, 0, stream>>>(gemm_buf, ffn_dw_w, u16);

    // pout 1x1 + residual -> d_out: [192,512] x [512, 8*16384] (WMMA)
    {
        int waves = (CDIM >> 5) * (int)(npx >> 5);
        wmma_gemm_kernel<HIDP><<<waves/8, TB, 0, stream>>>(wpout16, u16, (float*)d_out, xres,
                                                           CDIM, CDIM, CDIM);
    }
}
